// ContrastiveLoss_38285338477143
// MI455X (gfx1250) — matrix operand order repaired and checked by reading, not compiled, not run on hardware
//
#include <hip/hip_runtime.h>
#include <hip/hip_bf16.h>

typedef __attribute__((ext_vector_type(16))) __bf16 v16bf;
typedef __attribute__((ext_vector_type(4)))  __bf16 v4bf;
typedef __attribute__((ext_vector_type(8)))  float  v8f;

#define BB 128
#define NN 512
#define DD 1024
#define CMARGIN 0.3f
#define SLOTS 32        // max j-band partial slots per anchor row

// ------------------------------------------------------------------
// Kernel A: inverse L2 norm per (b,n) row; optionally also writes the
// normalized row as bf16 (embN) for the WMMA GEMM. One wave per row.
// ------------------------------------------------------------------
__global__ void __launch_bounds__(256) knorm(const float* __restrict__ emb,
                                             float* __restrict__ inv_norm,
                                             __bf16* __restrict__ embN) {
    const int wave = threadIdx.x >> 5;
    const int lane = threadIdx.x & 31;
    const int row  = blockIdx.x * 8 + wave;            // 0 .. BB*NN-1
    const float4* p = reinterpret_cast<const float4*>(emb + (size_t)row * DD);
    float4 vals[8];
    float s = 0.f;
#pragma unroll
    for (int i = 0; i < 8; ++i) {                      // 8 * 32 * 4 = 1024 floats
        vals[i] = p[lane + i * 32];
        s += vals[i].x * vals[i].x + vals[i].y * vals[i].y +
             vals[i].z * vals[i].z + vals[i].w * vals[i].w;
    }
#pragma unroll
    for (int off = 16; off >= 1; off >>= 1) s += __shfl_xor(s, off);
    const float inr = rsqrtf(s);                       // all lanes hold it
    if (lane == 0) inv_norm[row] = inr;
    if (embN) {
        v4bf* q = reinterpret_cast<v4bf*>(embN + (size_t)row * DD);
#pragma unroll
        for (int i = 0; i < 8; ++i) {
            v4bf o;
            o[0] = (__bf16)(vals[i].x * inr);
            o[1] = (__bf16)(vals[i].y * inr);
            o[2] = (__bf16)(vals[i].z * inr);
            o[3] = (__bf16)(vals[i].w * inr);
            q[lane + i * 32] = o;
        }
    }
}

// ------------------------------------------------------------------
// Kernel B: per-batch stats (n_pos, n_neg) and positive-sum vector
// P_b[d] = sum_{n: label==1} emb[b,n,d] * inv_norm[b,n].  1 block / batch.
// ------------------------------------------------------------------
__global__ void __launch_bounds__(256) kpos(const float* __restrict__ emb,
                                            const int* __restrict__ labels,
                                            const float* __restrict__ inv_norm,
                                            float* __restrict__ P,
                                            int* __restrict__ stats) {
    const int b = blockIdx.x;
    __shared__ float w[NN];
    __shared__ int   s_np;
    if (threadIdx.x == 0) s_np = 0;
    __syncthreads();
    int local = 0;
    for (int n = (int)threadIdx.x; n < NN; n += 256) {
        int pos = (labels[b * NN + n] == 1) ? 1 : 0;
        local  += pos;
        w[n]    = pos ? inv_norm[b * NN + n] : 0.f;
    }
    atomicAdd(&s_np, local);
    __syncthreads();
    if (threadIdx.x == 0) { stats[2 * b] = s_np; stats[2 * b + 1] = NN - s_np; }

    const float* eb = emb + (size_t)b * NN * DD;
    const int d = threadIdx.x;
    float a0 = 0.f, a1 = 0.f, a2 = 0.f, a3 = 0.f;
    for (int n = 0; n < NN; ++n) {
        float wn = w[n];                               // uniform across block
        if (wn != 0.f) {
            const float* r = eb + (size_t)n * DD;
            a0 += wn * r[d];
            a1 += wn * r[d + 256];
            a2 += wn * r[d + 512];
            a3 += wn * r[d + 768];
        }
    }
    float* Pb = P + b * DD;
    Pb[d] = a0; Pb[d + 256] = a1; Pb[d + 512] = a2; Pb[d + 768] = a3;
}

// ------------------------------------------------------------------
// Kernel C: pos_sim[b,n] = (embN_n.P_b - embN_n.embN_n)/max(np-1,1)
// (1.0 when np<=1, matching reference incl. explicit diag subtraction).
// One wave per row.
// ------------------------------------------------------------------
__global__ void __launch_bounds__(256) kpossim(const float* __restrict__ emb,
                                               const float* __restrict__ inv_norm,
                                               const float* __restrict__ P,
                                               const int* __restrict__ stats,
                                               float* __restrict__ pos_sim) {
    const int wave = threadIdx.x >> 5;
    const int lane = threadIdx.x & 31;
    const int row  = blockIdx.x * 8 + wave;            // b*NN + n
    const int b    = row >> 9;
    const float* e  = emb + (size_t)row * DD;
    const float* Pb = P + b * DD;
    const float inr = inv_norm[row];
    float dotP = 0.f, self = 0.f;
    for (int i = lane; i < DD; i += 32) {
        float v = e[i] * inr;
        dotP += v * Pb[i];
        self += v * v;
    }
#pragma unroll
    for (int off = 16; off >= 1; off >>= 1) {
        dotP += __shfl_xor(dotP, off);
        self += __shfl_xor(self, off);
    }
    if (lane == 0) {
        int np = stats[2 * b];
        int dn = np - 1; if (dn < 1) dn = 1;
        pos_sim[row] = (np > 1) ? (dotP - self) / (float)dn : 1.0f;
    }
}

// ==================================================================
// Kernel D (main path): register-blocked bf16 WMMA sim GEMM.
// One wave computes a 64x32 block of sim (4x2 tiles of 16x16):
//   8 x v_wmma_f32_16x16x32_bf16 per k-step vs 12 b128 loads.
// Fragments straight from bf16 embN rows (ISA lane layout):
//   A (16x32): lane row = l16; elems 0-7  = K half*8+0..7,
//                              elems 8-15 = K 16+half*8+0..7  -> 2 x b128
//   B (32x16): lane col = l16; same K order as A               -> 2 x b128
// C/D: lane l, reg r -> i = r + 8*(l>>4), j = l&15.
// Epilogue writes deterministic per-(b,i,jband) partials (no atomics).
// ==================================================================
__global__ void __launch_bounds__(256) ksim_blk(const __bf16* __restrict__ embN,
                                                const int* __restrict__ labels,
                                                const float* __restrict__ pos_sim,
                                                float* __restrict__ part) {
    const int wave = threadIdx.x >> 5;
    const int lane = threadIdx.x & 31;
    const int job  = blockIdx.x * 8 + wave;            // 0..16383
    const int b    = job >> 7;                         // 128 jobs / batch
    const int rem  = job & 127;
    const int it   = rem >> 4;                         // 0..7   (64-row band)
    const int jt   = rem & 15;                         // 0..15  (32-col band)
    const int half = lane >> 4;
    const int l16  = lane & 15;
    const int i0   = it * 64;
    const int j0   = jt * 32;

    const __bf16* base = embN + (size_t)b * NN * DD;
    const uint4* pA[4];
    const uint4* pB[2];
#pragma unroll
    for (int ia = 0; ia < 4; ++ia)
        pA[ia] = reinterpret_cast<const uint4*>(base + (size_t)(i0 + ia * 16 + l16) * DD);
#pragma unroll
    for (int jb = 0; jb < 2; ++jb)
        pB[jb] = reinterpret_cast<const uint4*>(base + (size_t)(j0 + jb * 16 + l16) * DD);

    v8f c[4][2];
#pragma unroll
    for (int ia = 0; ia < 4; ++ia)
#pragma unroll
        for (int jb = 0; jb < 2; ++jb) c[ia][jb] = (v8f){};

    union Frag { uint4 u[2]; v16bf v; };
    for (int k8 = 0; k8 < DD / 8; k8 += 4) {           // 32 K per step
        Frag a[4], bm[2];
#pragma unroll
        for (int ia = 0; ia < 4; ++ia) {
            a[ia].u[0] = pA[ia][k8 + half];
            a[ia].u[1] = pA[ia][k8 + 2 + half];
        }
#pragma unroll
        for (int jb = 0; jb < 2; ++jb) {
            bm[jb].u[0] = pB[jb][k8 + half];
            bm[jb].u[1] = pB[jb][k8 + 2 + half];
        }
#pragma unroll
        for (int ia = 0; ia < 4; ++ia)
#pragma unroll
            for (int jb = 0; jb < 2; ++jb)
                c[ia][jb] = __builtin_amdgcn_wmma_f32_16x16x32_bf16(
                    false, a[ia].v, false, bm[jb].v, (short)0, c[ia][jb], false, false);
    }

    // Fused epilogue: relu(sim - pos_sim_i + margin) * negf_j, sum over the
    // 32 columns of this band, write one partial per (b, i, jt).
    float negf0 = (labels[b * NN + j0 + l16]      == 0) ? 1.0f : 0.0f;
    float negf1 = (labels[b * NN + j0 + 16 + l16] == 0) ? 1.0f : 0.0f;
#pragma unroll
    for (int ia = 0; ia < 4; ++ia) {
        float acc[8];
#pragma unroll
        for (int r = 0; r < 8; ++r) {
            const int i = i0 + ia * 16 + r + half * 8;
            const float ps = pos_sim[b * NN + i];
            float v0 = c[ia][0][r] - ps + CMARGIN; v0 = v0 > 0.f ? v0 : 0.f;
            float v1 = c[ia][1][r] - ps + CMARGIN; v1 = v1 > 0.f ? v1 : 0.f;
            acc[r] = v0 * negf0 + v1 * negf1;
        }
#pragma unroll
        for (int off = 1; off < 16; off <<= 1)         // reduce 16-lane halves
#pragma unroll
            for (int r = 0; r < 8; ++r) acc[r] += __shfl_xor(acc[r], off);
        if (l16 == 0) {
#pragma unroll
            for (int r = 0; r < 8; ++r) {
                const int i = i0 + ia * 16 + r + half * 8;
                part[(size_t)(b * NN + i) * SLOTS + jt] = acc[r];
            }
        }
    }
}

// ------------------------------------------------------------------
// Kernel D (fallback when ws too small for bf16 embN): on-the-fly
// normalize+convert, one 16x16 tile per wave. Same partials layout
// (32 j-bands of 16 columns).
// ------------------------------------------------------------------
__global__ void __launch_bounds__(256) ksim_fly(const float* __restrict__ emb,
                                                const int* __restrict__ labels,
                                                const float* __restrict__ inv_norm,
                                                const float* __restrict__ pos_sim,
                                                float* __restrict__ part) {
    const int wave = threadIdx.x >> 5;
    const int lane = threadIdx.x & 31;
    const int tile = blockIdx.x * 8 + wave;            // 0 .. 131071
    const int b    = tile >> 10;
    const int rem  = tile & 1023;
    const int it   = rem >> 5;
    const int jt   = rem & 31;
    const int half = lane >> 4;
    const int l16  = lane & 15;

    const int rowA = it * 16 + l16;
    const int rowB = jt * 16 + l16;
    const float* __restrict__ pa = emb + (size_t)(b * NN + rowA) * DD;
    const float* __restrict__ pb = emb + (size_t)(b * NN + rowB) * DD;
    const float ina = inv_norm[b * NN + rowA];
    const float inb = inv_norm[b * NN + rowB];

    int kA[8], kB[8];
#pragma unroll
    for (int v = 0; v < 8; ++v) {
        kA[v] = ((v < 4) ? 0 : 16) + half * 8 + (v & 3) * 2;
        kB[v] = ((v < 4) ? 0 : 16) + half * 8 + (v & 3) * 2;
    }

    v8f c = {};
    for (int k0 = 0; k0 < DD; k0 += 32) {
        v16bf a, bmat;
#pragma unroll
        for (int v = 0; v < 8; ++v) {
            float2 fa = *reinterpret_cast<const float2*>(pa + k0 + kA[v]);
            float2 fb = *reinterpret_cast<const float2*>(pb + k0 + kB[v]);
            a[2 * v]        = (__bf16)(fa.x * ina);
            a[2 * v + 1]    = (__bf16)(fa.y * ina);
            bmat[2 * v]     = (__bf16)(fb.x * inb);
            bmat[2 * v + 1] = (__bf16)(fb.y * inb);
        }
        c = __builtin_amdgcn_wmma_f32_16x16x32_bf16(
                false, a, false, bmat, (short)0, c, false, false);
    }

    const float negf = (labels[b * NN + jt * 16 + l16] == 0) ? 1.0f : 0.0f;
    float acc[8];
#pragma unroll
    for (int r = 0; r < 8; ++r) {
        const int i = it * 16 + r + half * 8;
        float v = c[r] - pos_sim[b * NN + i] + CMARGIN;
        v = v > 0.f ? v : 0.f;
        acc[r] = v * negf;
    }
#pragma unroll
    for (int off = 1; off < 16; off <<= 1)
#pragma unroll
        for (int r = 0; r < 8; ++r) acc[r] += __shfl_xor(acc[r], off);
    if (l16 == 0) {
#pragma unroll
        for (int r = 0; r < 8; ++r) {
            const int i = it * 16 + r + half * 8;
            part[(size_t)(b * NN + i) * SLOTS + jt] = acc[r];
        }
    }
}

// ------------------------------------------------------------------
// Kernel E: deterministic final reduction -> scalar loss.
// ------------------------------------------------------------------
__global__ void __launch_bounds__(256) kfinal(const int* __restrict__ labels,
                                              const int* __restrict__ stats,
                                              const float* __restrict__ part,
                                              int nslots,
                                              float* __restrict__ out) {
    __shared__ float st[256];
    __shared__ float sc[256];
    float t = 0.f, cnt = 0.f;
    for (int idx = (int)threadIdx.x; idx < BB * NN; idx += 256) {
        const int b  = idx >> 9;
        const int np = stats[2 * b];
        const int nn = stats[2 * b + 1];
        const bool valid = (np > 0) && (nn > 0);
        if (valid && labels[idx] == 1) {
            float s = 0.f;
            const float* p = part + (size_t)idx * SLOTS;
            for (int k = 0; k < nslots; ++k) s += p[k];   // fixed order
            int dn = nn; if (dn < 1) dn = 1;
            t   += s / (float)dn;
            cnt += 1.f;
        }
    }
    st[threadIdx.x] = t; sc[threadIdx.x] = cnt;
    __syncthreads();
    for (int s = 128; s > 0; s >>= 1) {
        if ((int)threadIdx.x < s) {
            st[threadIdx.x] += st[threadIdx.x + s];
            sc[threadIdx.x] += sc[threadIdx.x + s];
        }
        __syncthreads();
    }
    if (threadIdx.x == 0) out[0] = st[0] / fmaxf(sc[0], 1.0f);
}

// ------------------------------------------------------------------
extern "C" void kernel_launch(void* const* d_in, const int* in_sizes, int n_in,
                              void* d_out, int out_size, void* d_ws, size_t ws_size,
                              hipStream_t stream) {
    (void)in_sizes; (void)n_in; (void)out_size;
    const float* emb    = (const float*)d_in[0];   // [128,512,1024] f32
    const int*   labels = (const int*)d_in[1];     // [128,512] int

    // Workspace layout (fp32 region ~9.4 MB, then optional bf16 embN 134 MB):
    float* ws       = (float*)d_ws;
    float* inv_norm = ws;                                    // BB*NN
    float* P        = inv_norm + BB * NN;                    // BB*DD
    float* pos_sim  = P + BB * DD;                           // BB*NN
    float* part     = pos_sim + BB * NN;                     // BB*NN*SLOTS
    int*   stats    = (int*)(part + (size_t)BB * NN * SLOTS); // 2*BB ints
    size_t f32_bytes = (size_t)((char*)(stats + 2 * BB) - (char*)d_ws);
    size_t embN_off  = (f32_bytes + 255) & ~(size_t)255;
    size_t need_full = embN_off + (size_t)BB * NN * DD * 2;
    const bool big   = (ws_size >= need_full);
    __bf16* embN = big ? (__bf16*)((char*)d_ws + embN_off) : nullptr;

    float* out = (float*)d_out;

    knorm  <<<BB * NN / 8, 256, 0, stream>>>(emb, inv_norm, embN);
    kpos   <<<BB,          256, 0, stream>>>(emb, labels, inv_norm, P, stats);
    kpossim<<<BB * NN / 8, 256, 0, stream>>>(emb, inv_norm, P, stats, pos_sim);
    if (big) {
        // 16384 wave-jobs (64x32 block each), 8 waves per workgroup
        ksim_blk<<<2048, 256, 0, stream>>>(embN, labels, pos_sim, part);
        kfinal  <<<1,    256, 0, stream>>>(labels, stats, part, 16, out);
    } else {
        ksim_fly<<<(BB * 32 * 32) / 8, 256, 0, stream>>>(emb, labels, inv_norm, pos_sim, part);
        kfinal  <<<1,    256, 0, stream>>>(labels, stats, part, 32, out);
    }
}
